// STaRNet_54417235641069
// MI455X (gfx1250) — matrix operand order repaired and checked by reading, not compiled, not run on hardware
//
#include <hip/hip_runtime.h>
#include <hip/hip_bf16.h>

// ---------------------------------------------------------------------------
// STaRNet pipeline for MI455X (gfx1250, wave32, WMMA).
// Heavy op: batched SYRK cov = h h^T (52 GFLOP) via v_wmma_f32_16x16x32_bf16,
// 64x64 register-blocked per wave (16 accumulators, 32 flop/byte from L2),
// upper-triangular tiles only (symmetry) with mirrored stores.
// ---------------------------------------------------------------------------

typedef __attribute__((ext_vector_type(16))) __bf16 v16bf;
typedef __attribute__((ext_vector_type(8)))  float  v8f;

union ABFrag { v16bf v; unsigned u[8]; __bf16 b[16]; };

#define BATCH 128
#define CIN   22
#define TT    1000     // input temporal length
#define TPOUT 1001     // temporal length after 'same-ish' depthwise convs
#define DCH   450      // h channels
#define HPAD  512      // 450 padded to 8 tiles of 64 for the SYRK
#define DPAD  464      // 450 padded to 29*16 for the W-map GEMM
#define KH    1024     // 1001 padded K for SYRK
#define KC    512      // 450 padded K for W-map GEMMs
#define NT64  8        // 64-row tiles in SYRK
#define NTRI  36       // NT64*(NT64+1)/2 upper-triangular tiles

__device__ __forceinline__ float bn1(float x, float g, float be, float mn, float vr) {
  return (x - mn) * (g * rsqrtf(vr + 1e-5f)) + be;
}

// ---------------- Stage 1: spatial convs + BN -> xs (B,60,1000) -------------
__global__ void k_spatial(const float* __restrict__ x,
                          const float* __restrict__ w1, const float* __restrict__ w3,
                          const float* __restrict__ w5,
                          const float* __restrict__ g, const float* __restrict__ be,
                          const float* __restrict__ mn, const float* __restrict__ vr,
                          float* __restrict__ xs) {
  int idx = blockIdx.x * blockDim.x + threadIdx.x;
  if (idx >= BATCH * 60 * TT) return;
  int t = idx % TT, r = (idx / TT) % 60, b = idx / (60 * TT);
  int br, c0, k; const float* w;
  if (r < 22)      { br = 0; c0 = r;      k = 1; w = w1; }
  else if (r < 42) { br = 1; c0 = r - 22; k = 3; w = w3; }
  else             { br = 2; c0 = r - 42; k = 5; w = w5; }
  const float* xb = x + (size_t)b * CIN * TT;
  float acc = 0.f;
  for (int j = 0; j < k; ++j) acc += w[j] * xb[(c0 + j) * TT + t];
  xs[idx] = bn1(acc, g[br], be[br], mn[br], vr[br]);
}

// ---------------- Stage 2: 50x60 feature mix + BN -> xf (B,50,1000) ---------
__global__ void k_feature(const float* __restrict__ xs, const float* __restrict__ wf,
                          const float* __restrict__ g, const float* __restrict__ be,
                          const float* __restrict__ mn, const float* __restrict__ vr,
                          float* __restrict__ xf) {
  int idx = blockIdx.x * blockDim.x + threadIdx.x;
  if (idx >= BATCH * 50 * TT) return;
  int t = idx % TT, f = (idx / TT) % 50, b = idx / (50 * TT);
  const float* col = xs + (size_t)b * 60 * TT + t;
  const float* wrow = wf + f * 60;
  float acc = 0.f;
  for (int r = 0; r < 60; ++r) acc += wrow[r] * col[r * TT];
  xf[idx] = bn1(acc, g[f], be[f], mn[f], vr[f]);
}

// -------- Stage 3: depthwise temporal convs + BN -> h bf16 (B,512,1024) -----
__global__ void k_temporal(const float* __restrict__ xf,
                           const float* __restrict__ wt0, const float* __restrict__ wt1,
                           const float* __restrict__ wt2,
                           const float* __restrict__ g, const float* __restrict__ be,
                           const float* __restrict__ mn, const float* __restrict__ vr,
                           __bf16* __restrict__ hbf) {
  int idx = blockIdx.x * blockDim.x + threadIdx.x;
  if (idx >= BATCH * HPAD * KH) return;
  int t = idx % KH, c = (idx / KH) % HPAD, b = idx / (HPAD * KH);
  if (c >= DCH || t >= TPOUT) { hbf[idx] = (__bf16)0.f; return; }
  int br, k; const float* w;
  if (c < 150)      { br = 0; k = 64; w = wt0; }
  else if (c < 300) { br = 1; k = 32; w = wt1; }
  else              { br = 2; k = 16; w = wt2; }
  int o = c - br * 150;
  int gch = o / 3;
  int pad = k >> 1;
  const float* row = xf + ((size_t)b * 50 + gch) * TT;
  float acc = 0.f;
  for (int j = 0; j < k; ++j) {
    int tt = t - pad + j;
    if (tt >= 0 && tt < TT) acc += row[tt] * w[o * k + j];
  }
  hbf[idx] = (__bf16)bn1(acc, g[c], be[c], mn[c], vr[c]);
}

// -------- Stage 4: batched SYRK via WMMA: cov = h h^T / 999 -----------------
// One wave computes a 64x64 output tile (16 16x16 WMMA accumulators).
// Per K-step: 4 A-frags + 4 B-frags (8KB wave-wide) feed 16 WMMAs -> 32 flop/B.
// Only upper-triangular tiles are computed; results mirrored on store.
__device__ __forceinline__ void load_afrag(ABFrag& a, const __bf16* row, int hl) {
  const unsigned* ap = (const unsigned*)row;
  int a0 = hl * 8;
  a.u[0] = ap[a0];     a.u[1] = ap[a0 + 1]; a.u[2] = ap[a0 + 2];  a.u[3] = ap[a0 + 3];
  a.u[4] = ap[a0 + 4]; a.u[5] = ap[a0 + 5]; a.u[6] = ap[a0 + 6]; a.u[7] = ap[a0 + 7];
}
__device__ __forceinline__ void load_bfrag_tr(ABFrag& bb, const __bf16* row, int hl) {
  const unsigned* bp = (const unsigned*)row;
  int b0 = hl * 8;
  bb.u[0] = bp[b0];     bb.u[1] = bp[b0 + 1]; bb.u[2] = bp[b0 + 2]; bb.u[3] = bp[b0 + 3];
  bb.u[4] = bp[b0 + 4]; bb.u[5] = bp[b0 + 5]; bb.u[6] = bp[b0 + 6]; bb.u[7] = bp[b0 + 7];
}

__global__ void __launch_bounds__(256) k_syrk(const __bf16* __restrict__ hbf,
                                              float* __restrict__ cov) {
  int wave = threadIdx.x >> 5;
  int tile = blockIdx.x * 8 + wave;
  if (tile >= NTRI) return;
  int b = blockIdx.y;
  // decode upper-triangular tile index -> (ti, tj), tj >= ti
  int t = tile, ti = 0;
  while (t >= NT64 - ti) { t -= NT64 - ti; ++ti; }
  int tj = ti + t;
  int lane = threadIdx.x & 31, hl = lane >> 4, l15 = lane & 15;
  const __bf16* H = hbf + (size_t)b * HPAD * KH;
  const __bf16* Ar0 = H + (size_t)(ti * 64 + l15) * KH;
  const __bf16* Br0 = H + (size_t)(tj * 64 + l15) * KH;
  v8f acc[16];
  for (int i = 0; i < 16; ++i) acc[i] = (v8f){};
  for (int kk = 0; kk < KH / 32; ++kk) {
    ABFrag af[4], bf[4];
    for (int i = 0; i < 4; ++i) load_afrag(af[i], Ar0 + (size_t)(i * 16) * KH + kk * 32, hl);
    for (int j = 0; j < 4; ++j) load_bfrag_tr(bf[j], Br0 + (size_t)(j * 16) * KH + kk * 32, hl);
    for (int i = 0; i < 4; ++i)
      for (int j = 0; j < 4; ++j)
        acc[i * 4 + j] = __builtin_amdgcn_wmma_f32_16x16x32_bf16(
            false, af[i].v, false, bf[j].v, (short)0, acc[i * 4 + j], false, false);
  }
  const float inv = 1.0f / 999.0f;
  float* Cb = cov + (size_t)b * DCH * DCH;
  for (int i = 0; i < 4; ++i)
    for (int j = 0; j < 4; ++j)
      for (int v = 0; v < 8; ++v) {
        int m = ti * 64 + i * 16 + v + hl * 8;
        int n = tj * 64 + j * 16 + l15;
        if (m < DCH && n < DCH) {
          float val = acc[i * 4 + j][v] * inv;
          Cb[(size_t)m * DCH + n] = val;
          Cb[(size_t)n * DCH + m] = val;   // symmetry (identical value on diag tiles)
        }
      }
}

// -------- Stage 5: shrinkage -----------------------------------------------
__global__ void k_diagmu(const float* __restrict__ cov, float* __restrict__ mu) {
  __shared__ float red[256];
  int b = blockIdx.x, tid = threadIdx.x;
  const float* Cb = cov + (size_t)b * DCH * DCH;
  float s = 0.f;
  for (int i = tid; i < DCH; i += 256) s += Cb[(size_t)i * DCH + i];
  red[tid] = s;
  __syncthreads();
  for (int st = 128; st > 0; st >>= 1) {
    if (tid < st) red[tid] += red[tid + st];
    __syncthreads();
  }
  if (tid == 0) mu[b] = red[0] / (float)DCH;
}

__global__ void k_shrink_bf(const float* __restrict__ cov, const float* __restrict__ mu,
                            __bf16* __restrict__ covbf) {
  int idx = blockIdx.x * blockDim.x + threadIdx.x;
  if (idx >= BATCH * DPAD * KC) return;
  int j = idx % KC, i = (idx / KC) % DPAD, b = idx / (DPAD * KC);
  float v = 0.f;
  if (i < DCH && j < DCH) {
    v = 0.95f * cov[(size_t)b * DCH * DCH + (size_t)i * DCH + j];
    if (i == j) v += 0.05f * mu[b];
  }
  covbf[idx] = (__bf16)v;
}

// -------- Stage 6: bf16 copies of W (K-padded) and W^T ----------------------
__global__ void k_prepW(const float* __restrict__ W, __bf16* __restrict__ Wbf,
                        __bf16* __restrict__ Wt) {
  int idx = blockIdx.x * blockDim.x + threadIdx.x;
  if (idx >= KC * 64) return;
  int k = idx / 64, n = idx % 64;
  float v = (k < DCH) ? W[k * 64 + n] : 0.f;
  Wbf[idx] = (__bf16)v;
  Wt[n * KC + k] = (__bf16)v;
}

// -------- Generic wave-tile WMMA GEMM: C = A(MxK,row) * B(KxN,row) ----------
__global__ void __launch_bounds__(256) k_gemm(const __bf16* __restrict__ A, int lda, long aStr,
                                              const __bf16* __restrict__ Bm, int ldb, long bStr,
                                              float* __restrict__ Cm, int ldc, long cStr,
                                              int mT, int nT, int ksteps, int mMax, int nMax) {
  int wave = threadIdx.x >> 5;
  int tile = blockIdx.x * 8 + wave;
  if (tile >= mT * nT) return;
  int b = blockIdx.y;
  int ti = tile / nT, tj = tile % nT;
  int lane = threadIdx.x & 31, hl = lane >> 4, l15 = lane & 15;
  const __bf16* Arow = A + (size_t)b * aStr + (size_t)(ti * 16 + l15) * lda;
  const __bf16* Bb = Bm + (size_t)b * bStr;
  int ncol = tj * 16 + l15;
  v8f acc = {};
  for (int kk = 0; kk < ksteps; ++kk) {
    ABFrag a, bb;
    load_afrag(a, Arow + kk * 32, hl);
    int kb = kk * 32 + hl * 16;
    for (int v = 0; v < 8; ++v) {
      int k0 = kb + 2 * v;
      bb.b[2 * v]     = Bb[(size_t)k0 * ldb + ncol];
      bb.b[2 * v + 1] = Bb[(size_t)(k0 + 1) * ldb + ncol];
    }
    acc = __builtin_amdgcn_wmma_f32_16x16x32_bf16(false, a.v, false, bb.v,
                                                  (short)0, acc, false, false);
  }
  float* Cb = Cm + (size_t)b * cStr;
  for (int v = 0; v < 8; ++v) {
    int m = ti * 16 + v + hl * 8;
    if (m < mMax && ncol < nMax) Cb[(size_t)m * ldc + ncol] = acc[v];
  }
}

// -------- Stage 7: tmp(f32, 464x64) -> bf16 K-padded (512x64) ---------------
__global__ void k_tmpbf(const float* __restrict__ tmp, __bf16* __restrict__ tmpbf) {
  int idx = blockIdx.x * blockDim.x + threadIdx.x;
  if (idx >= BATCH * KC * 64) return;
  int n = idx % 64, k = (idx / 64) % KC, b = idx / (KC * 64);
  float v = (k < DPAD) ? tmp[(size_t)b * DPAD * 64 + (size_t)k * 64 + n] : 0.f;
  tmpbf[idx] = (__bf16)v;
}

// -------- Stage 8: 64x64 eigh (parallel cyclic Jacobi) + logm + triu + FC ---
__global__ void __launch_bounds__(64) k_eig_fc(const float* __restrict__ mapped,
                                               const float* __restrict__ fcw,
                                               const float* __restrict__ fcb,
                                               float* __restrict__ out) {
  __shared__ float sA[64 * 65];   // row-padded to kill bank conflicts
  __shared__ float sV[64 * 65];
  __shared__ float sLam[64];
  __shared__ int   sP[32], sQ[32];
  __shared__ float sC[32], sSn[32];
  __shared__ float sRed[4 * 64];
  int tid = threadIdx.x, b = blockIdx.x;
  const float* M = mapped + (size_t)b * 4096;
  for (int j = 0; j < 64; ++j) {
    sA[tid * 65 + j] = 0.5f * (M[tid * 64 + j] + M[j * 64 + tid]);  // symmetrize
    sV[tid * 65 + j] = (tid == j) ? 1.f : 0.f;
  }
  __syncthreads();
  for (int sweep = 0; sweep < 10; ++sweep) {
    for (int r = 0; r < 63; ++r) {
      if (tid < 32) {  // 32 disjoint (p,q) pairs: round-robin circle method
        int p, q;
        if (tid == 0) { p = r; q = 63; }
        else {
          int a1 = (r + tid) % 63, a2 = (r + 63 - tid) % 63;
          p = a1 < a2 ? a1 : a2; q = a1 < a2 ? a2 : a1;
        }
        float apq = sA[p * 65 + q];
        float c = 1.f, sn = 0.f;
        if (fabsf(apq) > 1e-12f) {
          float tau = (sA[q * 65 + q] - sA[p * 65 + p]) / (2.f * apq);
          float tt = ((tau >= 0.f) ? 1.f : -1.f) / (fabsf(tau) + sqrtf(1.f + tau * tau));
          c = 1.f / sqrtf(1.f + tt * tt);
          sn = tt * c;
        }
        sP[tid] = p; sQ[tid] = q; sC[tid] = c; sSn[tid] = sn;
      }
      __syncthreads();
      for (int s2 = 0; s2 < 32; ++s2) {  // column rotations (A*J, V*J); row = tid
        int p = sP[s2], q = sQ[s2]; float c = sC[s2], sn = sSn[s2];
        float ap = sA[tid * 65 + p], aq = sA[tid * 65 + q];
        sA[tid * 65 + p] = c * ap - sn * aq; sA[tid * 65 + q] = sn * ap + c * aq;
        float vp = sV[tid * 65 + p], vq = sV[tid * 65 + q];
        sV[tid * 65 + p] = c * vp - sn * vq; sV[tid * 65 + q] = sn * vp + c * vq;
      }
      __syncthreads();
      for (int s2 = 0; s2 < 32; ++s2) {  // row rotations (J^T * A); col = tid
        int p = sP[s2], q = sQ[s2]; float c = sC[s2], sn = sSn[s2];
        float ap = sA[p * 65 + tid], aq = sA[q * 65 + tid];
        sA[p * 65 + tid] = c * ap - sn * aq; sA[q * 65 + tid] = sn * ap + c * aq;
      }
      __syncthreads();
    }
  }
  sLam[tid] = logf(fmaxf(sA[tid * 65 + tid], 1e-6f));
  __syncthreads();
  for (int j = 0; j < 64; ++j) {  // log-matrix: L = V diag(log lam) V^T
    float a = 0.f;
    for (int m = 0; m < 64; ++m) a += sV[tid * 65 + m] * sLam[m] * sV[j * 65 + m];
    sA[tid * 65 + j] = a;  // thread writes only its own row; reads only sV/sLam
  }
  __syncthreads();
  float a0 = 0.f, a1 = 0.f, a2 = 0.f, a3 = 0.f;
  int f = 0;
  for (int i = 0; i < 64; ++i)
    for (int j = i; j < 64; ++j) {
      if ((f & 63) == tid) {
        float v = sA[i * 65 + j];
        a0 += v * fcw[f]; a1 += v * fcw[2080 + f];
        a2 += v * fcw[4160 + f]; a3 += v * fcw[6240 + f];
      }
      ++f;
    }
  sRed[tid] = a0; sRed[64 + tid] = a1; sRed[128 + tid] = a2; sRed[192 + tid] = a3;
  __syncthreads();
  if (tid < 4) {
    float s = 0.f;
    for (int k = 0; k < 64; ++k) s += sRed[tid * 64 + k];
    out[b * 4 + tid] = s + fcb[tid];
  }
}

// ---------------------------------------------------------------------------
extern "C" void kernel_launch(void* const* d_in, const int* in_sizes, int n_in,
                              void* d_out, int out_size, void* d_ws, size_t ws_size,
                              hipStream_t stream) {
  const float* x        = (const float*)d_in[0];
  const float* ws0      = (const float*)d_in[1];
  const float* ws1      = (const float*)d_in[2];
  const float* ws2      = (const float*)d_in[3];
  const float* bns_g    = (const float*)d_in[4];
  const float* bns_b    = (const float*)d_in[5];
  const float* bns_m    = (const float*)d_in[6];
  const float* bns_v    = (const float*)d_in[7];
  const float* wf       = (const float*)d_in[8];
  const float* bnf_g    = (const float*)d_in[9];
  const float* bnf_b    = (const float*)d_in[10];
  const float* bnf_m    = (const float*)d_in[11];
  const float* bnf_v    = (const float*)d_in[12];
  const float* wt0      = (const float*)d_in[13];
  const float* wt1      = (const float*)d_in[14];
  const float* wt2      = (const float*)d_in[15];
  const float* bnt_g    = (const float*)d_in[16];
  const float* bnt_b    = (const float*)d_in[17];
  const float* bnt_m    = (const float*)d_in[18];
  const float* bnt_v    = (const float*)d_in[19];
  const float* W_map    = (const float*)d_in[20];
  const float* fc_w     = (const float*)d_in[21];
  const float* fc_b     = (const float*)d_in[22];
  float* out = (float*)d_out;

  // workspace layout (256B aligned)
  char* ws = (char*)d_ws;
  size_t off = 0;
  auto alloc = [&](size_t bytes) { void* p = ws + off; off = (off + bytes + 255) & ~(size_t)255; return p; };
  float*  xs     = (float*)alloc((size_t)BATCH * 60 * TT * 4);
  float*  xf     = (float*)alloc((size_t)BATCH * 50 * TT * 4);
  __bf16* hbf    = (__bf16*)alloc((size_t)BATCH * HPAD * KH * 2);
  float*  cov    = (float*)alloc((size_t)BATCH * DCH * DCH * 4);
  float*  mu     = (float*)alloc((size_t)BATCH * 4);
  __bf16* covbf  = (__bf16*)alloc((size_t)BATCH * DPAD * KC * 2);
  __bf16* Wbf    = (__bf16*)alloc((size_t)KC * 64 * 2);
  __bf16* Wtbf   = (__bf16*)alloc((size_t)64 * KC * 2);
  float*  tmp    = (float*)alloc((size_t)BATCH * DPAD * 64 * 4);
  __bf16* tmpbf  = (__bf16*)alloc((size_t)BATCH * KC * 64 * 2);
  float*  mapped = (float*)alloc((size_t)BATCH * 64 * 64 * 4);
  (void)ws_size; (void)in_sizes; (void)n_in; (void)out_size;

  int n1 = BATCH * 60 * TT;
  k_spatial<<<(n1 + 255) / 256, 256, 0, stream>>>(x, ws0, ws1, ws2, bns_g, bns_b, bns_m, bns_v, xs);

  int n2 = BATCH * 50 * TT;
  k_feature<<<(n2 + 255) / 256, 256, 0, stream>>>(xs, wf, bnf_g, bnf_b, bnf_m, bnf_v, xf);

  int n3 = BATCH * HPAD * KH;
  k_temporal<<<(n3 + 255) / 256, 256, 0, stream>>>(xf, wt0, wt1, wt2, bnt_g, bnt_b, bnt_m, bnt_v, hbf);

  k_syrk<<<dim3((NTRI + 7) / 8, BATCH), 256, 0, stream>>>(hbf, cov);

  k_diagmu<<<BATCH, 256, 0, stream>>>(cov, mu);

  int n6 = BATCH * DPAD * KC;
  k_shrink_bf<<<(n6 + 255) / 256, 256, 0, stream>>>(cov, mu, covbf);

  k_prepW<<<(KC * 64 + 255) / 256, 256, 0, stream>>>(W_map, Wbf, Wtbf);

  // tmp = covbf(464x512) * Wbf(512x64)
  k_gemm<<<dim3((29 * 4 + 7) / 8, BATCH), 256, 0, stream>>>(
      covbf, KC, (long)DPAD * KC, Wbf, 64, 0L, tmp, 64, (long)DPAD * 64,
      29, 4, KC / 32, DPAD, 64);

  int n9 = BATCH * KC * 64;
  k_tmpbf<<<(n9 + 255) / 256, 256, 0, stream>>>(tmp, tmpbf);

  // mapped = Wtbf(64x512) * tmpbf(512x64)
  k_gemm<<<dim3((4 * 4 + 7) / 8, BATCH), 256, 0, stream>>>(
      Wtbf, KC, 0L, tmpbf, 64, (long)KC * 64, mapped, 64, 4096L,
      4, 4, KC / 32, 64, 64);

  k_eig_fc<<<BATCH, 64, 0, stream>>>(mapped, fc_w, fc_b, out);
}
